// CausalSelfAttention_3100966388360
// MI455X (gfx1250) — matrix operand order repaired and checked by reading, not compiled, not run on hardware
//
#include <hip/hip_runtime.h>

typedef __bf16 bf16;
typedef __attribute__((ext_vector_type(16))) __bf16 v16bf;
typedef __attribute__((ext_vector_type(8)))  float  v8f;

static __device__ inline v8f wmma_bf16(v16bf a, v16bf b, v8f c) {
  return __builtin_amdgcn_wmma_f32_16x16x32_bf16(false, a, false, b, (short)0, c, false, false);
}

// CDNA5 async global->LDS copy (ASYNCcnt-tracked), 16B per lane.
// lds = 32-bit LDS byte address (low 32 bits of generic shared pointer),
// ga  = 64-bit global address.
#define ASYNC_B128(lds, ga)                                                   \
  asm volatile("global_load_async_to_lds_b128 %0, %1, off"                    \
               :: "v"(lds), "v"(ga) : "memory")
#define WAIT_ASYNC() asm volatile("s_wait_asynccnt 0x0" ::: "memory")

// ---------------------------------------------------------------- cast f32 -> bf16
__global__ __launch_bounds__(256) void cast_f2b(const float* __restrict__ x,
                                                bf16* __restrict__ y, int n) {
  int i = (blockIdx.x * 256 + threadIdx.x) * 4;
  if (i < n) {
    float4 v = *(const float4*)(x + i);
    alignas(8) bf16 t[4];
    t[0] = (bf16)v.x; t[1] = (bf16)v.y; t[2] = (bf16)v.z; t[3] = (bf16)v.w;
    *(uint2*)(y + i) = *(uint2*)t;
  }
}

// ---------------------------------------------------- transpose-cast W[K][N] -> Wt[N][K]
__global__ __launch_bounds__(256) void tcast(const float* __restrict__ w,
                                             bf16* __restrict__ wt, int K, int N) {
  int i = blockIdx.x * 256 + threadIdx.x;   // index into wt, K fastest
  if (i < N * K) {
    int k = i % K, n = i / K;
    wt[i] = (bf16)w[(size_t)k * N + n];
  }
}

// ---------------------------------------------------------------- WMMA GEMM
// C[M][N] = A[M][K] * Bt[N][K]^T  (A,Bt bf16; out fp32 or bf16)
// block tile 128x256, BK=32, double-buffered async LDS staging.
// 8 waves: wave (w&1) -> 64-row M half, (w>>1) -> 64-col N quarter.
// Per wave per k-step: 16 x v_wmma_f32_16x16x32_bf16, 16 x ds_load_b128.
template<bool OUT_BF16>
__global__ __launch_bounds__(256) void gemm_bf16(const bf16* __restrict__ A,
                                                 const bf16* __restrict__ Bt,
                                                 void* __restrict__ outv,
                                                 int M, int N, int K) {
  __shared__ bf16 As[2][128 * 40];   // row stride 40 (80B): conflict-free b128
  __shared__ bf16 Bs[2][256 * 40];
  const int tid  = threadIdx.x;
  const int w    = tid >> 5, lane = tid & 31;
  const int ln   = lane & 15, hi = lane >> 4;
  const int g8   = hi ? 8 : 0;     // unused
  const int g16  = hi ? 16 : 0;    // B-frag K group
  const int bm = blockIdx.x * 128, bn = blockIdx.y * 256;
  const int waveM = (w & 1) * 64, waveN = (w >> 1) * 64;
  const int arow = tid >> 1, acol = (tid & 1) * 16;
  const bf16* aptr = A  + (size_t)(bm + arow) * K + acol;
  const bf16* bptr = Bt + (size_t)(bn + tid) * K;

  // stage one BK=32 tile pair into buffer `buf` with async copies
  auto stage = [&](int buf, int kk) {
    unsigned int la = (unsigned int)(size_t)&As[buf][arow * 40 + acol];
    unsigned long long ga = (unsigned long long)(size_t)(aptr + kk);
    ASYNC_B128(la,      ga);
    ASYNC_B128(la + 16, ga + 16);
    unsigned int lb = (unsigned int)(size_t)&Bs[buf][tid * 40];
    unsigned long long gb = (unsigned long long)(size_t)(bptr + kk);
    ASYNC_B128(lb,      gb);
    ASYNC_B128(lb + 16, gb + 16);
    ASYNC_B128(lb + 32, gb + 32);
    ASYNC_B128(lb + 48, gb + 48);
  };

  v8f acc[4][4] = {};
  stage(0, 0);
  WAIT_ASYNC();
  __syncthreads();

  int buf = 0;
  for (int kk = 0; kk < K; kk += 32) {
    if (kk + 32 < K) stage(buf ^ 1, kk + 32);   // overlap next copy with WMMAs

    v16bf af[4], bfr[4];
#pragma unroll
    for (int mt = 0; mt < 4; mt++) {
      const bf16* ar = &As[buf][(waveM + mt * 16 + ln) * 40 + g16];
      ((uint4*)&af[mt])[0] = *(const uint4*)(ar);        // K g16..g16+7
      ((uint4*)&af[mt])[1] = *(const uint4*)(ar + 8);    // K g16+8..g16+15
    }
#pragma unroll
    for (int nt = 0; nt < 4; nt++) {
      const bf16* br = &Bs[buf][(waveN + nt * 16 + ln) * 40 + g16];
      ((uint4*)&bfr[nt])[0] = *(const uint4*)(br);       // K g16..g16+7
      ((uint4*)&bfr[nt])[1] = *(const uint4*)(br + 8);   // K g16+8..g16+15
    }
#pragma unroll
    for (int mt = 0; mt < 4; mt++)
#pragma unroll
      for (int nt = 0; nt < 4; nt++)
        acc[mt][nt] = wmma_bf16(af[mt], bfr[nt], acc[mt][nt]);

    WAIT_ASYNC();
    __syncthreads();
    buf ^= 1;
  }

#pragma unroll
  for (int mt = 0; mt < 4; mt++)
#pragma unroll
    for (int nt = 0; nt < 4; nt++) {
      int col = bn + waveN + nt * 16 + ln;
#pragma unroll
      for (int r = 0; r < 8; r++) {
        int row = bm + waveM + mt * 16 + r + 8 * hi;
        if (OUT_BF16) ((bf16*)outv)[(size_t)row * N + col] = (bf16)acc[mt][nt][r];
        else          ((float*)outv)[(size_t)row * N + col] = acc[mt][nt][r];
      }
    }
}

// ---------------------------------------------------------------- RoPE in-place on Q,K
// qkv bf16 [B*T][3072]; one thread per (b,t,h,i<32)
__global__ __launch_bounds__(256) void rope_kernel(bf16* __restrict__ qkv) {
  int idx = blockIdx.x * 256 + threadIdx.x;
  int i = idx & 31;
  int h = (idx >> 5) & 15;
  int t = (idx >> 9) & 2047;
  int b = idx >> 20;
  float freq = __expf(-(float)(2 * i) * (9.210340371976184f / 64.0f)); // 10000^{-2i/64}
  float th = (float)t * freq;
  float c = cosf(th), s = sinf(th);
  size_t base = ((size_t)(b * 2048 + t)) * 3072 + h * 64;
#pragma unroll
  for (int qk = 0; qk < 2; qk++) {            // q at +0, k at +1024
    size_t o = base + qk * 1024;
    float x1 = (float)qkv[o + i], x2 = (float)qkv[o + 32 + i];
    qkv[o + i]      = (bf16)(x1 * c - x2 * s);
    qkv[o + 32 + i] = (bf16)(x2 * c + x1 * s);
  }
}

// ---------------------------------------------------------------- flash attention
// grid = B*H*(T/128); block 256 (8 waves); wave w -> q rows [qbase+16w, +16)
__global__ __launch_bounds__(256) void flash_kernel(const bf16* __restrict__ qkv,
                                                    bf16* __restrict__ attn) {
  __shared__ bf16 Ks[32 * 72];      // K chunk rows [32][64], stride 72
  __shared__ bf16 Vt[64 * 40];      // V chunk transposed [d=64][k=32], stride 40
  __shared__ bf16 Ps[8][16 * 40];   // per-wave P tile [16][32], stride 40
  const int tid = threadIdx.x;
  const int w = tid >> 5, lane = tid & 31, ln = lane & 15, hi = lane >> 4;
  const int g8 = hi ? 8 : 0, g16 = hi ? 16 : 0;
  const int bid = blockIdx.x;
  const int qt = bid & 15, h = (bid >> 4) & 15, b = bid >> 8;
  const int qbase = qt * 128, wbase = qbase + w * 16;
  const size_t RS = 3072;

  // Q fragments (A layout, d in two 32-chunks), RoPE already applied
  const bf16* qptr = qkv + ((size_t)(b * 2048 + wbase + ln)) * RS + h * 64;
  v16bf qf0, qf1;
  ((uint4*)&qf0)[0] = *(const uint4*)(qptr + g16);
  ((uint4*)&qf0)[1] = *(const uint4*)(qptr + 8 + g16);
  ((uint4*)&qf1)[0] = *(const uint4*)(qptr + 32 + g16);
  ((uint4*)&qf1)[1] = *(const uint4*)(qptr + 40 + g16);

  v8f O0 = {}, O1 = {}, O2 = {}, O3 = {};
  float m[8], l[8];
#pragma unroll
  for (int r = 0; r < 8; r++) { m[r] = -1e30f; l[r] = 0.f; }

  const bf16* kbase = qkv + ((size_t)b * 2048) * RS + 1024 + h * 64;
  const bf16* vbase = qkv + ((size_t)b * 2048) * RS + 2048 + h * 64;
  const int krow = tid >> 3, kcol = (tid & 7) * 8;

  for (int kc = 0; kc < qbase + 128; kc += 32) {
    __syncthreads();
    // stage K chunk row-major via CDNA5 async LDS copy
    {
      unsigned int lk = (unsigned int)(size_t)&Ks[krow * 72 + kcol];
      unsigned long long gk =
          (unsigned long long)(size_t)(kbase + (size_t)(kc + krow) * RS + kcol);
      ASYNC_B128(lk, gk);
    }
    // stage V chunk transposed (element scatter, must go through VGPRs)
    uint4 vv = *(const uint4*)(vbase + (size_t)(kc + krow) * RS + kcol);
    const bf16* vx = (const bf16*)&vv;
#pragma unroll
    for (int j = 0; j < 8; j++) Vt[(kcol + j) * 40 + krow] = vx[j];
    WAIT_ASYNC();
    __syncthreads();

    if (kc <= wbase + 15) {                 // wave-uniform causal skip
      // S = Q * K^T  (two 16-key tiles)
      v16bf kf;
      v8f S0 = {}, S1 = {};
      {
        const bf16* kr = &Ks[ln * 72];
        ((uint4*)&kf)[0] = *(const uint4*)(kr + g16);
        ((uint4*)&kf)[1] = *(const uint4*)(kr + g16 + 8);
        S0 = wmma_bf16(qf0, kf, S0);
        ((uint4*)&kf)[0] = *(const uint4*)(kr + 32 + g16);
        ((uint4*)&kf)[1] = *(const uint4*)(kr + 40 + g16);
        S0 = wmma_bf16(qf1, kf, S0);
      }
      {
        const bf16* kr = &Ks[(16 + ln) * 72];
        ((uint4*)&kf)[0] = *(const uint4*)(kr + g16);
        ((uint4*)&kf)[1] = *(const uint4*)(kr + g16 + 8);
        S1 = wmma_bf16(qf0, kf, S1);
        ((uint4*)&kf)[0] = *(const uint4*)(kr + 32 + g16);
        ((uint4*)&kf)[1] = *(const uint4*)(kr + 40 + g16);
        S1 = wmma_bf16(qf1, kf, S1);
      }
      // online softmax in C-layout (lane = column, vgpr = row)
      float p0[8], p1[8];
      const int col0 = kc + ln, col1 = col0 + 16;
#pragma unroll
      for (int r = 0; r < 8; r++) {
        const int row = wbase + r + 8 * hi;
        float s0 = S0[r] * 0.125f; if (col0 > row) s0 = -1e9f;
        float s1 = S1[r] * 0.125f; if (col1 > row) s1 = -1e9f;
        float t = fmaxf(s0, s1);
        t = fmaxf(t, __shfl_xor(t, 1, 32));
        t = fmaxf(t, __shfl_xor(t, 2, 32));
        t = fmaxf(t, __shfl_xor(t, 4, 32));
        t = fmaxf(t, __shfl_xor(t, 8, 32));
        const float mn = fmaxf(m[r], t);
        const float alpha = __expf(m[r] - mn);
        p0[r] = __expf(s0 - mn);
        p1[r] = __expf(s1 - mn);
        float rs = p0[r] + p1[r];
        rs += __shfl_xor(rs, 1, 32);
        rs += __shfl_xor(rs, 2, 32);
        rs += __shfl_xor(rs, 4, 32);
        rs += __shfl_xor(rs, 8, 32);
        l[r] = l[r] * alpha + rs;
        m[r] = mn;
        O0[r] *= alpha; O1[r] *= alpha; O2[r] *= alpha; O3[r] *= alpha;
      }
      // C-layout -> A-layout for P via per-wave LDS scratch
      bf16* pw = &Ps[w][0];
#pragma unroll
      for (int r = 0; r < 8; r++) {
        pw[(r + 8 * hi) * 40 + ln]      = (bf16)p0[r];
        pw[(r + 8 * hi) * 40 + 16 + ln] = (bf16)p1[r];
      }
      v16bf pf;
      const bf16* pr = &Ps[w][ln * 40 + g16];
      ((uint4*)&pf)[0] = *(const uint4*)(pr);
      ((uint4*)&pf)[1] = *(const uint4*)(pr + 8);
      // O += P * V
      {
        v16bf vf;
        const bf16* vr;
        vr = &Vt[(0 * 16 + ln) * 40 + g16];
        ((uint4*)&vf)[0] = *(const uint4*)(vr); ((uint4*)&vf)[1] = *(const uint4*)(vr + 8);
        O0 = wmma_bf16(pf, vf, O0);
        vr = &Vt[(1 * 16 + ln) * 40 + g16];
        ((uint4*)&vf)[0] = *(const uint4*)(vr); ((uint4*)&vf)[1] = *(const uint4*)(vr + 8);
        O1 = wmma_bf16(pf, vf, O1);
        vr = &Vt[(2 * 16 + ln) * 40 + g16];
        ((uint4*)&vf)[0] = *(const uint4*)(vr); ((uint4*)&vf)[1] = *(const uint4*)(vr + 8);
        O2 = wmma_bf16(pf, vf, O2);
        vr = &Vt[(3 * 16 + ln) * 40 + g16];
        ((uint4*)&vf)[0] = *(const uint4*)(vr); ((uint4*)&vf)[1] = *(const uint4*)(vr + 8);
        O3 = wmma_bf16(pf, vf, O3);
      }
    }
  }
  // normalize and write bf16 attn-out [B*T][1024]
  bf16* outp = attn + ((size_t)(b * 2048 + wbase)) * 1024 + h * 64;
#pragma unroll
  for (int r = 0; r < 8; r++) {
    const float inv = 1.0f / l[r];
    const int row = r + 8 * hi;
    outp[(size_t)row * 1024 +  0 + ln] = (bf16)(O0[r] * inv);
    outp[(size_t)row * 1024 + 16 + ln] = (bf16)(O1[r] * inv);
    outp[(size_t)row * 1024 + 32 + ln] = (bf16)(O2[r] * inv);
    outp[(size_t)row * 1024 + 48 + ln] = (bf16)(O3[r] * inv);
  }
}

// ---------------------------------------------------------------- launcher
extern "C" void kernel_launch(void* const* d_in, const int* in_sizes, int n_in,
                              void* d_out, int out_size, void* d_ws, size_t ws_size,
                              hipStream_t stream) {
  const float* x     = (const float*)d_in[0];   // [4,2048,1024]
  const float* w_qkv = (const float*)d_in[1];   // [1024,3072]
  const float* w_out = (const float*)d_in[2];   // [1024,1024]
  float* out = (float*)d_out;

  char* p = (char*)d_ws;
  bf16* xb    = (bf16*)p; p += (size_t)8192 * 1024 * 2;   // x bf16 [8192][1024]
  bf16* wqkvt = (bf16*)p; p += (size_t)3072 * 1024 * 2;   // w_qkv^T bf16 [3072][1024]
  bf16* woutt = (bf16*)p; p += (size_t)1024 * 1024 * 2;   // w_out^T bf16 [1024][1024]
  bf16* qkv   = (bf16*)p; p += (size_t)8192 * 3072 * 2;   // qkv bf16 [8192][3072]
  bf16* attn  = (bf16*)p; p += (size_t)8192 * 1024 * 2;   // attn-out bf16 [8192][1024]

  cast_f2b<<<8192, 256, 0, stream>>>(x, xb, 8192 * 1024);
  tcast<<<12288, 256, 0, stream>>>(w_qkv, wqkvt, 1024, 3072);
  tcast<<<4096, 256, 0, stream>>>(w_out, woutt, 1024, 1024);
  gemm_bf16<true><<<dim3(64, 12), 256, 0, stream>>>(xb, wqkvt, (void*)qkv, 8192, 3072, 1024);
  rope_kernel<<<16384, 256, 0, stream>>>(qkv);
  flash_kernel<<<1024, 256, 0, stream>>>(qkv, attn);
  gemm_bf16<false><<<dim3(64, 4), 256, 0, stream>>>(attn, woutt, (void*)out, 8192, 1024, 1024);
}
